// MultiHeadAttentionLayer_73126113181805
// MI455X (gfx1250) — compile-verified, matrix operand order checked by reading, not run on hardware
//
#include <hip/hip_runtime.h>
#include <hip/hip_bf16.h>

#define B_  4
#define S_  2048
#define D_  512
#define H_  8
#define DH_ 64

typedef __attribute__((ext_vector_type(16))) _Float16 v16h;
typedef __attribute__((ext_vector_type(8)))  float    v8f;
typedef _Float16 half_t;

// -------- WMMA helpers (wave32, 16x16x32 f16 -> f32) --------

__device__ __forceinline__ v8f wmma_f16(v16h a, v16h b, v8f c) {
    // (neg_a, A, neg_b, B, c_mod, C, reuse_a, reuse_b)
    return __builtin_amdgcn_wmma_f32_16x16x32_f16(false, a, false, b, (short)0, c,
                                                  false, false);
}

// Load a 16x32 f16 fragment. Works for A ([M][K] row-major, lane->row)
// and for B when B is stored [N][K] (lane->col). K-pairs are contiguous,
// so each of the 8 VGPRs is a single dword load.
__device__ __forceinline__ v16h ld_frag_h(const half_t* base, int stride,
                                          int row0, int k0) {
    int l = threadIdx.x & 31;
    int m = l & 15;
    int g = l >> 4;
    const half_t* p = base + (size_t)(row0 + m) * stride + (k0 + g * 8);
    union { v16h h; unsigned int u[8]; } f;
#pragma unroll
    for (int r = 0; r < 8; ++r) {
        int k = ((r >> 2) << 4) + ((r & 3) << 1);
        f.u[r] = *(const unsigned int*)(p + k);
    }
    return f.h;
}

// Same fragment but sourced from f32 (attn matrix), converted on load.
__device__ __forceinline__ v16h ld_frag_f32(const float* base, int stride,
                                            int row0, int k0) {
    int l = threadIdx.x & 31;
    int m = l & 15;
    int g = l >> 4;
    const float* p = base + (size_t)(row0 + m) * stride + (k0 + g * 8);
    v16h h;
#pragma unroll
    for (int r = 0; r < 8; ++r) {
        int k = ((r >> 2) << 4) + ((r & 3) << 1);
        float2 t = *(const float2*)(p + k);
        h[2 * r]     = (_Float16)t.x;
        h[2 * r + 1] = (_Float16)t.y;
    }
    return h;
}

__device__ __forceinline__ v8f vzero8() {
    v8f z = {0.f, 0.f, 0.f, 0.f, 0.f, 0.f, 0.f, 0.f};
    return z;
}

// -------- Precision conversion kernels --------

__global__ void cvt_f32_to_f16_kernel(const float* __restrict__ x,
                                      half_t* __restrict__ y, int n) {
    int i = blockIdx.x * blockDim.x + threadIdx.x;
    if (i < n) y[i] = (_Float16)x[i];
}

// w [K][N] f32 row-major -> wt [N][K] f16 (so B-frag k-pairs are contiguous)
__global__ void transpose_cvt_kernel(const float* __restrict__ w,
                                     half_t* __restrict__ wt) {
    int idx = blockIdx.x * blockDim.x + threadIdx.x;  // idx = n*D_ + k
    if (idx < D_ * D_) {
        int n = idx >> 9;
        int k = idx & (D_ - 1);
        wt[idx] = (_Float16)w[k * D_ + n];
    }
}

// -------- Projection GEMM: C[8192x512] = A[8192x512] * Bt^T + bias --------
// mode 0: f16 out, head-split  [B,H,S,DH]           (Q, K)
// mode 1: f16 out, V-transposed [B,H,DH,S]          (V)
// mode 2: f32 out, straight    [B*S, D]             (final projection)
__global__ void proj_gemm_kernel(const half_t* __restrict__ A,
                                 const half_t* __restrict__ Bt,
                                 const float* __restrict__ bias,
                                 half_t* __restrict__ out16,
                                 float* __restrict__ out32, int mode) {
    const int wid = threadIdx.x >> 5;
    const int wm = wid & 3, wn = wid >> 2;
    const int m0 = blockIdx.x * 128 + wm * 32;
    const int n0 = blockIdx.y * 128 + wn * 64;

    v8f acc[2][4];
#pragma unroll
    for (int mi = 0; mi < 2; ++mi)
#pragma unroll
        for (int ni = 0; ni < 4; ++ni) acc[mi][ni] = vzero8();

    for (int k0 = 0; k0 < D_; k0 += 32) {
        v16h a0 = ld_frag_h(A, D_, m0, k0);
        v16h a1 = ld_frag_h(A, D_, m0 + 16, k0);
#pragma unroll
        for (int ni = 0; ni < 4; ++ni) {
            v16h b = ld_frag_h(Bt, D_, n0 + ni * 16, k0);
            acc[0][ni] = wmma_f16(a0, b, acc[0][ni]);
            acc[1][ni] = wmma_f16(a1, b, acc[1][ni]);
        }
    }

    const int l = threadIdx.x & 31;
    const int nl = l & 15, g = l >> 4;
#pragma unroll
    for (int mi = 0; mi < 2; ++mi) {
#pragma unroll
        for (int ni = 0; ni < 4; ++ni) {
#pragma unroll
            for (int r = 0; r < 8; ++r) {
                int row = m0 + mi * 16 + r + 8 * g;
                int col = n0 + ni * 16 + nl;
                float v = acc[mi][ni][r] + bias[col];
                int b = row >> 11, s = row & (S_ - 1);
                int h = col >> 6, dh = col & (DH_ - 1);
                if (mode == 0) {
                    out16[((size_t)(b * H_ + h) * S_ + s) * DH_ + dh] = (_Float16)v;
                } else if (mode == 1) {
                    out16[((size_t)(b * H_ + h) * DH_ + dh) * S_ + s] = (_Float16)v;
                } else {
                    out32[(size_t)row * D_ + col] = v;
                }
            }
        }
    }
}

// -------- scores = scale * Q K^T, lower triangle only --------
__global__ void attn_scores_kernel(const half_t* __restrict__ Qh,
                                   const half_t* __restrict__ Kh,
                                   float* __restrict__ attn) {
    const int ib = blockIdx.x, jb = blockIdx.y, bh = blockIdx.z;
    if (jb > ib) return;  // fully masked: softmax pass writes the zeros

    const int wid = threadIdx.x >> 5;
    const int wm = wid & 3, wn = wid >> 2;
    const int i0 = ib * 128 + wm * 32;
    const int j0 = jb * 128 + wn * 64;
    if (j0 > i0 + 31) return;  // wave-uniform: whole wave tile masked

    const half_t* Q = Qh + (size_t)bh * S_ * DH_;
    const half_t* K = Kh + (size_t)bh * S_ * DH_;

    v8f acc[2][4];
#pragma unroll
    for (int mi = 0; mi < 2; ++mi)
#pragma unroll
        for (int ni = 0; ni < 4; ++ni) acc[mi][ni] = vzero8();

#pragma unroll
    for (int k0 = 0; k0 < DH_; k0 += 32) {
        v16h a0 = ld_frag_h(Q, DH_, i0, k0);
        v16h a1 = ld_frag_h(Q, DH_, i0 + 16, k0);
#pragma unroll
        for (int ni = 0; ni < 4; ++ni) {
            v16h b = ld_frag_h(K, DH_, j0 + ni * 16, k0);
            acc[0][ni] = wmma_f16(a0, b, acc[0][ni]);
            acc[1][ni] = wmma_f16(a1, b, acc[1][ni]);
        }
    }

    const float scale = 0.125f;  // 1/sqrt(64)
    const int l = threadIdx.x & 31;
    const int nl = l & 15, g = l >> 4;
    float* out = attn + (size_t)bh * S_ * S_;
#pragma unroll
    for (int mi = 0; mi < 2; ++mi) {
#pragma unroll
        for (int ni = 0; ni < 4; ++ni) {
#pragma unroll
            for (int r = 0; r < 8; ++r) {
                int i = i0 + mi * 16 + r + 8 * g;
                int j = j0 + ni * 16 + nl;
                if (j <= i) out[(size_t)i * S_ + j] = acc[mi][ni][r] * scale;
            }
        }
    }
}

// -------- in-place causal row softmax (one 256-thread block per row) --------
__global__ void softmax_rows_kernel(float* __restrict__ attn) {
    const size_t row = blockIdx.x;          // 0 .. B*H*S-1
    const int i = (int)(row & (S_ - 1));    // query position
    float* p = attn + row * (size_t)S_;
    const int tid = threadIdx.x;
    __shared__ float red[256];

    float m = -1e30f;
    for (int j = tid; j <= i; j += 256) m = fmaxf(m, p[j]);
    red[tid] = m;
    __syncthreads();
    for (int s = 128; s > 0; s >>= 1) {
        if (tid < s) red[tid] = fmaxf(red[tid], red[tid + s]);
        __syncthreads();
    }
    m = red[0];
    __syncthreads();

    float sum = 0.f;
    for (int j = tid; j <= i; j += 256) {
        float e = __expf(p[j] - m);
        p[j] = e;
        sum += e;
    }
    red[tid] = sum;
    __syncthreads();
    for (int s = 128; s > 0; s >>= 1) {
        if (tid < s) red[tid] += red[tid + s];
        __syncthreads();
    }
    float inv = 1.0f / red[0];

    for (int j = tid; j <= i; j += 256) p[j] *= inv;
    for (int j = i + 1 + tid; j < S_; j += 256) p[j] = 0.0f;  // masked region
}

// -------- ctx = attn * V   (per (b,h); causal: skip zero key-blocks) --------
__global__ void attn_ctx_kernel(const float* __restrict__ attn,
                                const half_t* __restrict__ Vt,
                                half_t* __restrict__ ctx16) {
    const int mb = blockIdx.x, bh = blockIdx.y;
    const int wid = threadIdx.x >> 5;
    const int m0 = mb * 128 + wid * 16;

    const float* A = attn + (size_t)bh * S_ * S_;
    const half_t* V = Vt + (size_t)bh * DH_ * S_;  // [DH][S]

    v8f acc[4];
#pragma unroll
    for (int ni = 0; ni < 4; ++ni) acc[ni] = vzero8();

    const int nk = (m0 + 16 + 31) >> 5;  // keys <= m0+15 contribute
    for (int kk = 0; kk < nk; ++kk) {
        int k0 = kk * 32;
        v16h a = ld_frag_f32(A, S_, m0, k0);
#pragma unroll
        for (int ni = 0; ni < 4; ++ni) {
            v16h b = ld_frag_h(V, S_, ni * 16, k0);  // lane->dh, vgpr->k pairs
            acc[ni] = wmma_f16(a, b, acc[ni]);
        }
    }

    const int l = threadIdx.x & 31;
    const int nl = l & 15, g = l >> 4;
    const int b = bh >> 3, h = bh & (H_ - 1);
#pragma unroll
    for (int ni = 0; ni < 4; ++ni) {
#pragma unroll
        for (int r = 0; r < 8; ++r) {
            int s = m0 + r + 8 * g;
            int dh = ni * 16 + nl;
            ctx16[((size_t)(b * S_ + s)) * D_ + h * DH_ + dh] = (_Float16)acc[ni][r];
        }
    }
}

// -------- host orchestration --------

extern "C" void kernel_launch(void* const* d_in, const int* in_sizes, int n_in,
                              void* d_out, int out_size, void* d_ws, size_t ws_size,
                              hipStream_t stream) {
    const float* q  = (const float*)d_in[0];
    const float* k  = (const float*)d_in[1];
    const float* v  = (const float*)d_in[2];
    // d_in[3] = mask: causality implemented structurally, not needed
    const float* wq = (const float*)d_in[4];
    const float* bq = (const float*)d_in[5];
    const float* wk = (const float*)d_in[6];
    const float* bk = (const float*)d_in[7];
    const float* wv = (const float*)d_in[8];
    const float* bv = (const float*)d_in[9];
    const float* wo = (const float*)d_in[10];
    const float* bo = (const float*)d_in[11];

    float* out  = (float*)d_out;                       // [B,S,D]
    float* attn = out + (size_t)B_ * S_ * D_;          // [B,H,S,S]

    const size_t BSD = (size_t)B_ * S_ * D_;           // 4,194,304
    const size_t DD  = (size_t)D_ * D_;                // 262,144

    half_t* w    = (half_t*)d_ws;
    half_t* q16  = w;
    half_t* k16  = q16 + BSD;
    half_t* v16  = k16 + BSD;
    half_t* wqT  = v16 + BSD;
    half_t* wkT  = wqT + DD;
    half_t* wvT  = wkT + DD;
    half_t* woT  = wvT + DD;
    half_t* Qh   = woT + DD;   // [B,H,S,DH]
    half_t* Kh   = Qh + BSD;   // [B,H,S,DH]
    half_t* Vt   = Kh + BSD;   // [B,H,DH,S]
    half_t* ctx  = Vt + BSD;   // [B,S,D] f16

    // 1) downconvert activations + transpose/downconvert weights
    cvt_f32_to_f16_kernel<<<(int)(BSD / 256), 256, 0, stream>>>(q, q16, (int)BSD);
    cvt_f32_to_f16_kernel<<<(int)(BSD / 256), 256, 0, stream>>>(k, k16, (int)BSD);
    cvt_f32_to_f16_kernel<<<(int)(BSD / 256), 256, 0, stream>>>(v, v16, (int)BSD);
    transpose_cvt_kernel<<<(int)(DD / 256), 256, 0, stream>>>(wq, wqT);
    transpose_cvt_kernel<<<(int)(DD / 256), 256, 0, stream>>>(wk, wkT);
    transpose_cvt_kernel<<<(int)(DD / 256), 256, 0, stream>>>(wv, wvT);
    transpose_cvt_kernel<<<(int)(DD / 256), 256, 0, stream>>>(wo, woT);

    // 2) QKV projections (WMMA)
    dim3 pg(64, 4);
    proj_gemm_kernel<<<pg, 256, 0, stream>>>(q16, wqT, bq, Qh, nullptr, 0);
    proj_gemm_kernel<<<pg, 256, 0, stream>>>(k16, wkT, bk, Kh, nullptr, 0);
    proj_gemm_kernel<<<pg, 256, 0, stream>>>(v16, wvT, bv, Vt, nullptr, 1);

    // 3) scores (lower triangle) -> attn region
    dim3 sg(S_ / 128, S_ / 128, B_ * H_);
    attn_scores_kernel<<<sg, 256, 0, stream>>>(Qh, Kh, attn);

    // 4) causal softmax in place (also zeros the masked upper triangle)
    softmax_rows_kernel<<<B_ * H_ * S_, 256, 0, stream>>>(attn);

    // 5) ctx = attn @ V (WMMA, causal block skipping)
    dim3 cg(S_ / 128, B_ * H_);
    attn_ctx_kernel<<<cg, 256, 0, stream>>>(attn, Vt, ctx);

    // 6) output projection -> d_out
    proj_gemm_kernel<<<pg, 256, 0, stream>>>(ctx, woT, bo, nullptr, out, 2);
}
